// MultiHeadSelfAttention2DOnFrame_85950885528320
// MI455X (gfx1250) — compile-verified, matrix operand order checked by reading, not run on hardware
//
#include <hip/hip_runtime.h>

// ---------------------------------------------------------------------------
// MultiHeadSelfAttention2DOnFrame for MI455X (gfx1250, wave32, WMMA bf16)
// ---------------------------------------------------------------------------

#define DI __device__ __forceinline__

typedef __attribute__((ext_vector_type(16))) __bf16 v16bf;
typedef __attribute__((ext_vector_type(8)))  __bf16 v8bf;
typedef __attribute__((ext_vector_type(8)))  float  v8f;
typedef __attribute__((ext_vector_type(4)))  int    v4i;

// Async global->LDS copy engine (ASYNCcnt), if this toolchain exposes it.
#if defined(__has_builtin)
#  if __has_builtin(__builtin_amdgcn_global_load_async_to_lds_b128) && \
      __has_builtin(__builtin_amdgcn_s_wait_asynccnt)
#    define USE_ASYNC_LDS 1
#  endif
#endif
#ifndef USE_ASYNC_LDS
#  define USE_ASYNC_LDS 0
#endif

#if USE_ASYNC_LDS
typedef __attribute__((address_space(1))) v4i* gv4p;  // global int4*
typedef __attribute__((address_space(3))) v4i* lv4p;  // LDS int4*
#endif

constexpr int B_  = 4;
constexpr int C_  = 128;
constexpr int T_  = 1000;
constexpr int TP  = 1024;   // padded T
constexpr int F_  = 128;
constexpr int NH_ = 4;
constexpr int HID_= 16;
constexpr int VD_ = 32;
constexpr int EQ  = HID_ * F_;  // 2048
constexpr int EV  = VD_  * F_;  // 4096
constexpr float EPS = 1e-5f;
constexpr float ATT_SCALE = 0.022097086912079608f; // 1/sqrt(2048)

// workspace layout (bytes); all offsets multiple of 16
constexpr size_t WQKV_OFF = 0;                                   // 256x128 bf16
constexpr size_t WP_OFF   = WQKV_OFF + (size_t)256*128*2;        // 128x128 bf16
constexpr size_t QF_OFF   = WP_OFF   + (size_t)128*128*2;        // [hb][TP][EQ] bf16
constexpr size_t KF_OFF   = QF_OFF   + (size_t)NH_*B_*TP*EQ*2;
constexpr size_t VF_OFF   = KF_OFF   + (size_t)NH_*B_*TP*EQ*2;   // [hb][TP][EV] bf16
constexpr size_t S_OFF    = VF_OFF   + (size_t)NH_*B_*TP*EV*2;   // [hb][TP][TP] f32
constexpr size_t P_OFF    = S_OFF    + (size_t)16*TP*TP*4;       // [hb][TP][TP] bf16
constexpr size_t O_OFF    = P_OFF    + (size_t)16*TP*TP*2;       // [B][C][TP][F] bf16

union Frag { v16bf v; v8bf h[2]; };

// Copy 16 contiguous bytes global -> LDS. Uses the CDNA5 async-to-LDS path
// (global_load_async_to_lds_b128, ASYNCcnt) when available, else a VGPR
// round-trip (global_load_b128 + ds_store_b128).
DI void copy16_g2l(const __bf16* g, __bf16* l) {
#if USE_ASYNC_LDS
  __builtin_amdgcn_global_load_async_to_lds_b128((gv4p)g, (lv4p)l, 0, 0);
#else
  *(v8bf*)l = *(const v8bf*)g;
#endif
}

DI void copy_join() {  // make this wave's async copies visible before barrier
#if USE_ASYNC_LDS
  __builtin_amdgcn_s_wait_asynccnt(0);
#endif
}

// Load a 16x32 bf16 operand fragment. Source tile stored row-major with the
// K dimension contiguous (leading dim = ld elements). Works for the A operand
// (rows = M) and for the B operand when B is staged transposed (rows = N),
// matching the gfx1250 16-bit A-matrix VGPR layout (ISA 7.12.2).
DI v16bf load_frag(const __bf16* base, int ld) {
  const int lane = threadIdx.x & 31;
  const __bf16* p = base + (lane & 15) * ld + (lane >> 4) * 8;
  Frag f;
  f.h[0] = *(const v8bf*)(p);
  f.h[1] = *(const v8bf*)(p + 16);
  return f.v;
}

DI v8f wmma_bf16(v16bf a, v16bf b, v8f c) {
  return __builtin_amdgcn_wmma_f32_16x16x32_bf16(
      /*neg_a=*/false, a, /*neg_b=*/false, b,
      /*c_mod=*/(short)0, c, /*reuse_a=*/false, /*reuse_b=*/false);
}

DI v8f vzero() {
  v8f z;
  #pragma unroll
  for (int i = 0; i < 8; ++i) z[i] = 0.f;
  return z;
}

DI float wave_sum(float v) {
  #pragma unroll
  for (int m = 16; m >= 1; m >>= 1) v += __shfl_xor(v, m, 32);
  return v;
}

DI float wave_max(float v) {
  #pragma unroll
  for (int m = 16; m >= 1; m >>= 1) v = fmaxf(v, __shfl_xor(v, m, 32));
  return v;
}

// ---------------------------------------------------------------------------
// Kernel 0: pack weights to bf16
// ---------------------------------------------------------------------------
__global__ void k_prep(const float* __restrict__ Wq, const float* __restrict__ Wk,
                       const float* __restrict__ Wv, const float* __restrict__ Wp,
                       char* __restrict__ ws) {
  const int i = blockIdx.x * 256 + threadIdx.x;
  __bf16* wqkv = (__bf16*)(ws + WQKV_OFF);
  __bf16* wp   = (__bf16*)(ws + WP_OFF);
  if (i < 256 * 128) {
    const int po = i >> 7, c = i & 127;
    const int h = po >> 6, r = po & 63;
    float v;
    if (r < 16)      v = Wq[((size_t)h * HID_ + r)        * C_ + c];
    else if (r < 32) v = Wk[((size_t)h * HID_ + (r - 16)) * C_ + c];
    else             v = Wv[((size_t)h * VD_  + (r - 32)) * C_ + c];
    wqkv[i] = (__bf16)v;
  } else if (i < 256 * 128 + 128 * 128) {
    const int j = i - 256 * 128;
    wp[j] = (__bf16)Wp[j];
  }
}

// ---------------------------------------------------------------------------
// Kernel 1: fused QKV projection + PReLU + LayerNorm4D, one block per (b,t)
// Packed output rows: head h -> rows [64h..64h+63] = 16 q | 16 k | 32 v.
// Wave w (of 8) owns rows 32w..32w+31: even wave = q+k of head w/2 (two LN
// segments), odd wave = v of head w/2 (one LN segment) -> stats are wave-local.
// ---------------------------------------------------------------------------
__global__ __launch_bounds__(256) void k_qkv(
    const float* __restrict__ x,
    const float* __restrict__ bq, const float* __restrict__ aq,
    const float* __restrict__ gq, const float* __restrict__ beq,
    const float* __restrict__ bk, const float* __restrict__ ak,
    const float* __restrict__ gk, const float* __restrict__ bek,
    const float* __restrict__ bv, const float* __restrict__ av,
    const float* __restrict__ gv, const float* __restrict__ bev,
    char* __restrict__ ws) {
  const int tid = threadIdx.x;
  const int bid = blockIdx.x;
  const int b = bid >> 10, t = bid & (TP - 1);

  __bf16* qf = (__bf16*)(ws + QF_OFF);
  __bf16* kf = (__bf16*)(ws + KF_OFF);
  __bf16* vf = (__bf16*)(ws + VF_OFF);
  const __bf16* wqkv = (const __bf16*)(ws + WQKV_OFF);

  if (t >= T_) {  // zero the padded time rows so all GEMM tiles are full
    const __bf16 z = (__bf16)0.0f;
    for (int i = tid; i < NH_ * EQ; i += 256) {
      const int h = i / EQ, e = i - h * EQ;
      qf[((size_t)(h * B_ + b) * TP + t) * EQ + e] = z;
      kf[((size_t)(h * B_ + b) * TP + t) * EQ + e] = z;
    }
    for (int i = tid; i < NH_ * EV; i += 256) {
      const int h = i / EV, e = i - h * EV;
      vf[((size_t)(h * B_ + b) * TP + t) * EV + e] = z;
    }
    return;
  }

  __shared__ __align__(16) __bf16 xt[F_ * C_];  // [f][c] (Bt layout)
  for (int i = tid; i < C_ * F_; i += 256) {
    const int c = i >> 7, f = i & 127;
    xt[f * C_ + c] = (__bf16)x[(((size_t)b * C_ + c) * T_ + t) * F_ + f];
  }
  __syncthreads();

  const int w = tid >> 5, lane = tid & 31;
  const int rowadd = (lane >> 4) * 8, coll = lane & 15;
  const int h = w >> 1;
  const int row0 = 32 * w;

  v8f acc[2][8];
  #pragma unroll
  for (int i = 0; i < 2; ++i) {
    #pragma unroll
    for (int n = 0; n < 8; ++n) acc[i][n] = vzero();
  }

  #pragma unroll
  for (int kk = 0; kk < 4; ++kk) {
    const v16bf a0 = load_frag(wqkv + (size_t)(row0)      * C_ + kk * 32, C_);
    const v16bf a1 = load_frag(wqkv + (size_t)(row0 + 16) * C_ + kk * 32, C_);
    #pragma unroll
    for (int n = 0; n < 8; ++n) {
      const v16bf bt = load_frag(xt + n * 16 * C_ + kk * 32, C_);
      acc[0][n] = wmma_bf16(a0, bt, acc[0][n]);
      acc[1][n] = wmma_bf16(a1, bt, acc[1][n]);
    }
  }

  // bias + PReLU, accumulate per-m-tile stats
  float s1[2] = {0.f, 0.f}, s2[2] = {0.f, 0.f};
  #pragma unroll
  for (int i = 0; i < 2; ++i) {
    #pragma unroll
    for (int n = 0; n < 8; ++n) {
      #pragma unroll
      for (int r = 0; r < 8; ++r) {
        const int rl = r + rowadd;  // 0..15 within m-tile
        float v = acc[i][n][r];
        if ((w & 1) == 0) {
          const float* bias = (i == 0) ? bq : bk;
          const float* al   = (i == 0) ? aq : ak;
          v += bias[h * HID_ + rl];
          const float a = al[h];
          v = (v >= 0.f) ? v : a * v;
        } else {
          const int ov = i * 16 + rl;  // 0..31 in V
          v += bv[h * VD_ + ov];
          const float a = av[h];
          v = (v >= 0.f) ? v : a * v;
        }
        acc[i][n][r] = v;
        s1[i] += v; s2[i] += v * v;
      }
    }
  }
  s1[0] = wave_sum(s1[0]); s2[0] = wave_sum(s2[0]);
  s1[1] = wave_sum(s1[1]); s2[1] = wave_sum(s2[1]);

  float mu[2], rs[2];
  if ((w & 1) == 0) {
    #pragma unroll
    for (int i = 0; i < 2; ++i) {
      const float m = s1[i] * (1.f / 2048.f);
      const float var = s2[i] * (1.f / 2048.f) - m * m;
      mu[i] = m; rs[i] = rsqrtf(var + EPS);
    }
  } else {
    const float m = (s1[0] + s1[1]) * (1.f / 4096.f);
    const float var = (s2[0] + s2[1]) * (1.f / 4096.f) - m * m;
    mu[0] = mu[1] = m; rs[0] = rs[1] = rsqrtf(var + EPS);
  }

  const int hb = h * B_ + b;
  #pragma unroll
  for (int i = 0; i < 2; ++i) {
    #pragma unroll
    for (int n = 0; n < 8; ++n) {
      #pragma unroll
      for (int r = 0; r < 8; ++r) {
        const int rl = r + rowadd;
        const int f = n * 16 + coll;
        float v = (acc[i][n][r] - mu[i]) * rs[i];
        if ((w & 1) == 0) {
          const float* g  = (i == 0) ? gq  : gk;
          const float* be = (i == 0) ? beq : bek;
          v = v * g[(h * HID_ + rl) * F_ + f] + be[(h * HID_ + rl) * F_ + f];
          __bf16* dst = (i == 0) ? qf : kf;
          dst[((size_t)hb * TP + t) * EQ + rl * F_ + f] = (__bf16)v;
        } else {
          const int ov = i * 16 + rl;
          v = v * gv[(h * VD_ + ov) * F_ + f] + bev[(h * VD_ + ov) * F_ + f];
          vf[((size_t)hb * TP + t) * EV + ov * F_ + f] = (__bf16)v;
        }
      }
    }
  }
}

// ---------------------------------------------------------------------------
// Kernel 2: S = (Qf . Kf^T) * scale  (128x128 tiles, causal tile skip)
// ---------------------------------------------------------------------------
__global__ __launch_bounds__(256) void k_scores(char* __restrict__ ws) {
  const int bx = blockIdx.x, by = blockIdx.y, hb = blockIdx.z;
  if (bx > by) return;  // strictly-upper tiles are never read by softmax
  const int t0 = by * 128, s0 = bx * 128;

  const __bf16* qf = (const __bf16*)(ws + QF_OFF) + (size_t)hb * TP * EQ;
  const __bf16* kf = (const __bf16*)(ws + KF_OFF) + (size_t)hb * TP * EQ;
  float* S = (float*)(ws + S_OFF) + (size_t)hb * TP * TP;

  __shared__ __align__(16) __bf16 Al[128 * 64];
  __shared__ __align__(16) __bf16 Bl[128 * 64];

  const int tid = threadIdx.x, w = tid >> 5, lane = tid & 31;
  const int rowadd = (lane >> 4) * 8, coll = lane & 15;

  v8f acc[8];
  #pragma unroll
  for (int n = 0; n < 8; ++n) acc[n] = vzero();

  for (int kc = 0; kc < EQ / 64; ++kc) {
    const int k0 = kc * 64;
    for (int i = tid; i < 128 * 8; i += 256) {  // 8 groups of 8 bf16 per row
      const int row = i >> 3, g = i & 7;
      copy16_g2l(qf + (size_t)(t0 + row) * EQ + k0 + g * 8, &Al[row * 64 + g * 8]);
      copy16_g2l(kf + (size_t)(s0 + row) * EQ + k0 + g * 8, &Bl[row * 64 + g * 8]);
    }
    if (kc + 1 < EQ / 64) {  // pull next K-chunk into cache (global_prefetch_b8)
      __builtin_prefetch(qf + (size_t)(t0 + (tid >> 1)) * EQ + k0 + 64, 0, 1);
      __builtin_prefetch(kf + (size_t)(s0 + (tid >> 1)) * EQ + k0 + 64, 0, 1);
    }
    copy_join();
    __syncthreads();
    #pragma unroll
    for (int kk = 0; kk < 2; ++kk) {
      const v16bf a = load_frag(Al + (16 * w) * 64 + kk * 32, 64);
      #pragma unroll
      for (int n = 0; n < 8; ++n) {
        const v16bf bt = load_frag(Bl + n * 16 * 64 + kk * 32, 64);
        acc[n] = wmma_bf16(a, bt, acc[n]);
      }
    }
    __syncthreads();
  }

  #pragma unroll
  for (int n = 0; n < 8; ++n) {
    #pragma unroll
    for (int r = 0; r < 8; ++r) {
      const int tt = t0 + 16 * w + r + rowadd;
      const int ss = s0 + 16 * n + coll;
      S[(size_t)tt * TP + ss] = acc[n][r] * ATT_SCALE;
    }
  }
}

// ---------------------------------------------------------------------------
// Kernel 3: causal softmax row-wise, S(f32) -> P(bf16), zero past diagonal
// ---------------------------------------------------------------------------
__global__ __launch_bounds__(256) void k_softmax(char* __restrict__ ws) {
  const int t = blockIdx.x, hb = blockIdx.y;
  const float* Srow = (const float*)(ws + S_OFF) + ((size_t)hb * TP + t) * TP;
  __bf16* Prow = (__bf16*)(ws + P_OFF) + ((size_t)hb * TP + t) * TP;

  const int tid = threadIdx.x, w = tid >> 5, lane = tid & 31;
  __shared__ float red[8];

  float m = -3.4e38f;
  for (int s = tid; s <= t; s += 256) m = fmaxf(m, Srow[s]);
  m = wave_max(m);
  if (lane == 0) red[w] = m;
  __syncthreads();
  float mall = red[0];
  #pragma unroll
  for (int j = 1; j < 8; ++j) mall = fmaxf(mall, red[j]);
  __syncthreads();

  float sum = 0.f;
  for (int s = tid; s <= t; s += 256) sum += __expf(Srow[s] - mall);
  sum = wave_sum(sum);
  if (lane == 0) red[w] = sum;
  __syncthreads();
  float sall = 0.f;
  #pragma unroll
  for (int j = 0; j < 8; ++j) sall += red[j];
  const float inv = 1.0f / sall;

  for (int s = tid; s < TP; s += 256) {
    const float p = (s <= t) ? __expf(Srow[s] - mall) * inv : 0.f;
    Prow[s] = (__bf16)p;
  }
}

// ---------------------------------------------------------------------------
// Kernel 4: O = P . Vf   (K-loop truncated causally), write [B][C][TP][F] bf16
// ---------------------------------------------------------------------------
__global__ __launch_bounds__(256) void k_pv(char* __restrict__ ws) {
  const int bx = blockIdx.x, by = blockIdx.y, hb = blockIdx.z;
  const int h = hb / B_, b = hb - h * B_;
  const int t0 = by * 128, e0 = bx * 128;

  const __bf16* P = (const __bf16*)(ws + P_OFF) + (size_t)hb * TP * TP;
  const __bf16* V = (const __bf16*)(ws + VF_OFF) + (size_t)hb * TP * EV;
  __bf16* O = (__bf16*)(ws + O_OFF);

  __shared__ __align__(16) __bf16 Al[128 * 64];  // P tile [t][s]
  __shared__ __align__(16) __bf16 Bl[128 * 64];  // V^T tile [e][s]

  const int tid = threadIdx.x, w = tid >> 5, lane = tid & 31;
  const int rowadd = (lane >> 4) * 8, coll = lane & 15;

  v8f acc[8];
  #pragma unroll
  for (int n = 0; n < 8; ++n) acc[n] = vzero();

  const int nch_full = TP / 64;
  const int nch_caus = 2 * by + 2;  // s <= t0+127 suffices (P zero above diag)
  const int nch = nch_caus < nch_full ? nch_caus : nch_full;

  for (int kc = 0; kc < nch; ++kc) {
    const int s0 = kc * 64;
    for (int i = tid; i < 128 * 8; i += 256) {  // P tile: contiguous, async path
      const int row = i >> 3, g = i & 7;
      copy16_g2l(P + (size_t)(t0 + row) * TP + s0 + g * 8, &Al[row * 64 + g * 8]);
    }
    for (int i = tid; i < 64 * 128; i += 256) {  // coalesced read, transposed store
      const int sl = i >> 7, el = i & 127;
      Bl[el * 64 + sl] = V[(size_t)(s0 + sl) * EV + e0 + el];
    }
    copy_join();
    __syncthreads();
    #pragma unroll
    for (int kk = 0; kk < 2; ++kk) {
      const v16bf a = load_frag(Al + (16 * w) * 64 + kk * 32, 64);
      #pragma unroll
      for (int n = 0; n < 8; ++n) {
        const v16bf bt = load_frag(Bl + n * 16 * 64 + kk * 32, 64);
        acc[n] = wmma_bf16(a, bt, acc[n]);
      }
    }
    __syncthreads();
  }

  const int c = h * VD_ + bx;  // e0 spans exactly one vdim slot: vo == bx
  #pragma unroll
  for (int n = 0; n < 8; ++n) {
    #pragma unroll
    for (int r = 0; r < 8; ++r) {
      const int tt = t0 + 16 * w + r + rowadd;
      const int f = 16 * n + coll;
      O[(((size_t)b * C_ + c) * TP + tt) * F_ + f] = (__bf16)acc[n][r];
    }
  }
}

// ---------------------------------------------------------------------------
// Kernel 5: output projection + PReLU + LayerNorm (block-wide) + residual
// ---------------------------------------------------------------------------
__global__ __launch_bounds__(256) void k_final(
    const float* __restrict__ x, const float* __restrict__ bp,
    const float* __restrict__ ap, const float* __restrict__ gp,
    const float* __restrict__ bep, float* __restrict__ out,
    char* __restrict__ ws) {
  const int bid = blockIdx.x;
  const int b = bid / T_, t = bid - b * T_;

  const __bf16* O  = (const __bf16*)(ws + O_OFF);
  const __bf16* wp = (const __bf16*)(ws + WP_OFF);

  __shared__ __align__(16) __bf16 ot[F_ * C_];  // [f][c]
  __shared__ float red[2][8];

  const int tid = threadIdx.x, w = tid >> 5, lane = tid & 31;
  const int rowadd = (lane >> 4) * 8, coll = lane & 15;

  for (int i = tid; i < C_ * F_; i += 256) {
    const int c = i >> 7, f = i & 127;
    ot[f * C_ + c] = O[(((size_t)b * C_ + c) * TP + t) * F_ + f];
  }
  __syncthreads();

  v8f acc[8];
  #pragma unroll
  for (int n = 0; n < 8; ++n) acc[n] = vzero();

  #pragma unroll
  for (int kk = 0; kk < 4; ++kk) {
    const v16bf a = load_frag(wp + (size_t)(16 * w) * C_ + kk * 32, C_);
    #pragma unroll
    for (int n = 0; n < 8; ++n) {
      const v16bf bt = load_frag(ot + n * 16 * C_ + kk * 32, C_);
      acc[n] = wmma_bf16(a, bt, acc[n]);
    }
  }

  const float alpha = ap[0];
  float s1 = 0.f, s2 = 0.f;
  #pragma unroll
  for (int n = 0; n < 8; ++n) {
    #pragma unroll
    for (int r = 0; r < 8; ++r) {
      const int o = 16 * w + r + rowadd;
      float v = acc[n][r] + bp[o];
      v = (v >= 0.f) ? v : alpha * v;
      acc[n][r] = v;
      s1 += v; s2 += v * v;
    }
  }
  s1 = wave_sum(s1); s2 = wave_sum(s2);
  if (lane == 0) { red[0][w] = s1; red[1][w] = s2; }
  __syncthreads();
  float S1 = 0.f, S2 = 0.f;
  #pragma unroll
  for (int j = 0; j < 8; ++j) { S1 += red[0][j]; S2 += red[1][j]; }
  const float mu = S1 * (1.f / 16384.f);
  const float rs = rsqrtf(S2 * (1.f / 16384.f) - mu * mu + EPS);

  #pragma unroll
  for (int n = 0; n < 8; ++n) {
    #pragma unroll
    for (int r = 0; r < 8; ++r) {
      const int o = 16 * w + r + rowadd;
      const int f = 16 * n + coll;
      float v = (acc[n][r] - mu) * rs;
      v = v * gp[o * F_ + f] + bep[o * F_ + f];
      const size_t idx = (((size_t)b * C_ + o) * T_ + t) * F_ + f;
      out[idx] = v + x[idx];
    }
  }
}

// ---------------------------------------------------------------------------
extern "C" void kernel_launch(void* const* d_in, const int* in_sizes, int n_in,
                              void* d_out, int out_size, void* d_ws, size_t ws_size,
                              hipStream_t stream) {
  const float* x   = (const float*)d_in[0];
  const float* Wq  = (const float*)d_in[1];
  const float* bq  = (const float*)d_in[2];
  const float* aq  = (const float*)d_in[3];
  const float* gq  = (const float*)d_in[4];
  const float* beq = (const float*)d_in[5];
  const float* Wk  = (const float*)d_in[6];
  const float* bk  = (const float*)d_in[7];
  const float* ak  = (const float*)d_in[8];
  const float* gk  = (const float*)d_in[9];
  const float* bek = (const float*)d_in[10];
  const float* Wv  = (const float*)d_in[11];
  const float* bv  = (const float*)d_in[12];
  const float* av  = (const float*)d_in[13];
  const float* gv  = (const float*)d_in[14];
  const float* bev = (const float*)d_in[15];
  const float* Wp  = (const float*)d_in[16];
  const float* bp  = (const float*)d_in[17];
  const float* ap  = (const float*)d_in[18];
  const float* gp  = (const float*)d_in[19];
  const float* bep = (const float*)d_in[20];
  float* out = (float*)d_out;
  char* ws = (char*)d_ws;

  k_prep<<<dim3((256 * 128 + 128 * 128 + 255) / 256), 256, 0, stream>>>(Wq, Wk, Wv, Wp, ws);
  k_qkv<<<dim3(B_ * TP), 256, 0, stream>>>(x, bq, aq, gq, beq, bk, ak, gk, bek,
                                           bv, av, gv, bev, ws);
  k_scores<<<dim3(TP / 128, TP / 128, NH_ * B_), 256, 0, stream>>>(ws);
  k_softmax<<<dim3(T_, NH_ * B_), 256, 0, stream>>>(ws);
  k_pv<<<dim3(EV / 128, TP / 128, NH_ * B_), 256, 0, stream>>>(ws);
  k_final<<<dim3(B_ * T_), 256, 0, stream>>>(x, bp, ap, gp, bep, out, ws);
}
